// SelfAttention_695784702662
// MI455X (gfx1250) — matrix operand order repaired and checked by reading, not compiled, not run on hardware
//
#include <hip/hip_runtime.h>

// Problem constants (B=2, L=2048, D=3072, H=24, Hd=128)
#define BDIM 3072
#define SEQ  2048
#define NH   24
#define HD   128
#define BL   4096      // B*L
#define TD   9216      // 3*D

typedef __attribute__((ext_vector_type(16))) __bf16 v16bf;
typedef __attribute__((ext_vector_type(8)))  __bf16 v8bf;
typedef __attribute__((ext_vector_type(8)))  float  v8f;
typedef __attribute__((ext_vector_type(4)))  unsigned int v4u;
typedef __attribute__((ext_vector_type(8)))  int v8i;
typedef __attribute__((ext_vector_type(4)))  int v4i;

static __device__ __forceinline__ v8f wmma_bf16(v16bf a, v16bf b, v8f c) {
  return __builtin_amdgcn_wmma_f32_16x16x32_bf16(false, a, false, b, (short)0, c,
                                                 false, false);
}

union AFrag { v16bf v; v8bf h[2]; };

// ---------------------------------------------------------------------------
// TDM: 2-D tile load (global -> LDS), data_size = 2 bytes (bf16).
// D# layout per CDNA5 ISA. Groups 2/3 = 0 (2-D).
// 6-arg builtin: (uint32x4 g0, int32x8 g1, int32x4, int32x4, int32x8, i32 cpol)
static __device__ __forceinline__ void tdm_load_2d(unsigned lds_off,
                                                   const void* gptr,
                                                   unsigned tensor_d0,
                                                   unsigned tensor_d1,
                                                   unsigned long long stride0,
                                                   unsigned tile_d0,
                                                   unsigned tile_d1,
                                                   unsigned pad_bits) {
  unsigned long long ga = (unsigned long long)(size_t)gptr;
  v4u g0;
  g0[0] = 1u;                                            // count=1, user mode
  g0[1] = lds_off;                                       // LDS byte address
  g0[2] = (unsigned)(ga & 0xFFFFFFFFu);                  // global_addr[31:0]
  g0[3] = (unsigned)((ga >> 32) & 0x01FFFFFFu) | (2u << 30); // addr[56:32]|type=2
  v8i g1;
  g1[0] = (int)((1u << 16) | pad_bits);                  // data_size=2B (+pad)
  g1[1] = (int)((tensor_d0 & 0xFFFFu) << 16);            // tensor_dim0[15:0]
  g1[2] = (int)((tensor_d0 >> 16) | ((tensor_d1 & 0xFFFFu) << 16));
  g1[3] = (int)((tensor_d1 >> 16) | (tile_d0 << 16));    // tile_dim0
  g1[4] = (int)(tile_d1 & 0xFFFFu);                      // tile_dim1 (tile_dim2=0)
  g1[5] = (int)(stride0 & 0xFFFFFFFFull);                // tensor_dim0_stride lo
  g1[6] = (int)((stride0 >> 32) & 0xFFFFull);            // stride hi (dim1 stride=0)
  g1[7] = 0;
  v4i z4 = {0, 0, 0, 0};
  v8i z8 = {0, 0, 0, 0, 0, 0, 0, 0};
  __builtin_amdgcn_tensor_load_to_lds(g0, g1, z4, z4, z8, 0);
}

// pad configs: +16B (4 DWORDs, code 3) per row
#define PAD_ROW128B ((1u << 20) | (4u << 22) | (3u << 25))  // every 32 DW (128B row)
#define PAD_ROW256B ((1u << 20) | (5u << 22) | (3u << 25))  // every 64 DW (256B row)
#define STRIDE72  72   // 128B row + 16B pad, in bf16 elems
#define STRIDE136 136  // 256B row + 16B pad, in bf16 elems

// ---------------------------------------------------------------------------
// A-matrix fragment (16x32 bf16, row-major source):
// lane L: row = L&15 ; halves 0..7 = K kh..kh+7 ; halves 8..15 = K 16+kh..
static __device__ __forceinline__ v16bf load_a16(const __bf16* A, int lda,
                                                 int row0, int k0, int lane) {
  const int r  = lane & 15;
  const int kh = (lane >> 4) << 3;
  const __bf16* p = A + (size_t)(row0 + r) * lda + k0 + kh;
  AFrag f;
  f.h[0] = *(const v8bf*)p;
  f.h[1] = *(const v8bf*)(p + 16);
  return f.v;
}

// B-matrix fragment (32x16) from an LDS tile: lane = column(n) selects a tile
// row; halves 0..7 = K kh..kh+7, halves 8..15 = K 16+kh.. (two 16B loads).
static __device__ __forceinline__ v16bf lds_b_frag(const __bf16* base, int rs,
                                                   int col0, int k0, int lane) {
  const int n  = lane & 15;
  const int kh = (lane >> 4) << 3;
  const __bf16* p = base + (col0 + n) * rs + k0 + kh;
  AFrag f;
  f.h[0] = *(const v8bf*)p;
  f.h[1] = *(const v8bf*)(p + 16);
  return f.v;
}

static __device__ __forceinline__ v16bf lds_a_frag(const __bf16* base, int rs,
                                                   int row0, int k0, int lane) {
  const int r  = lane & 15;
  const int kh = (lane >> 4) << 3;
  const __bf16* p = base + (row0 + r) * rs + k0 + kh;
  AFrag f;
  f.h[0] = *(const v8bf*)p;
  f.h[1] = *(const v8bf*)(p + 16);
  return f.v;
}

// ---------------------------------------------------------------------------
__global__ void f32_to_bf16_kernel(const float* __restrict__ in,
                                   __bf16* __restrict__ out, int n) {
  int i = blockIdx.x * blockDim.x + threadIdx.x;
  if (i < n) out[i] = (__bf16)in[i];
}

// ---------------------------------------------------------------------------
// 8-WMMA strip with a depth-2 software pipeline (3 rotating B buffers): the
// fragment consumed by WMMA f was loaded 2 WMMAs earlier, so DS latency is
// covered without requiring the scheduler to keep 8 fragments live.
static __device__ __forceinline__ void wmma_strip8(const __bf16* tB, int rs,
                                                   int k0, int lane, v16bf a,
                                                   v8f acc[8]) {
  v16bf b0 = lds_b_frag(tB, rs, 0, k0, lane);
  v16bf b1 = lds_b_frag(tB, rs, 16, k0, lane);
  v16bf b2;
#pragma unroll
  for (int f = 0; f < 8; ++f) {
    if (f + 2 < 8) b2 = lds_b_frag(tB, rs, (f + 2) * 16, k0, lane);
    acc[f] = wmma_bf16(a, b0, acc[f]);
    b0 = b1;
    b1 = b2;
  }
}

// ---------------------------------------------------------------------------
// Shared GEMM mainloop: block tile 128x128, TDM double-buffered K-chunks of 64.
// Wave w owns rows [16w,16w+16). Result: acc[8] (16x128 strip per wave).
#define GEMM_TILE_B (128 * STRIDE72 * 2)  // 18432 bytes per tile

static __device__ __forceinline__ void gemm_mainloop_tdm(
    const __bf16* Ag, const __bf16* Bg, int K, char* smem, int wave, int lane,
    v8f acc[8]) {
  __bf16* lA = (__bf16*)smem;                       // [2][128*72]
  __bf16* lB = (__bf16*)(smem + 2 * GEMM_TILE_B);   // [2][128*72]
  const unsigned ldsA = (unsigned)(size_t)(void*)lA;
  const unsigned ldsB = (unsigned)(size_t)(void*)lB;
  const int NC = K / 64;

  if (wave == 0) {
    tdm_load_2d(ldsA, Ag, 64, 128, (unsigned long long)K, 64, 128, PAD_ROW128B);
    tdm_load_2d(ldsB, Bg, 64, 128, (unsigned long long)K, 64, 128, PAD_ROW128B);
  }
  for (int c = 0; c < NC; ++c) {
    const int buf = c & 1;
    if (wave == 0) __builtin_amdgcn_s_wait_tensorcnt((short)0);
    __syncthreads();
    if (wave == 0 && c + 1 < NC) {
      const int nb = buf ^ 1;
      tdm_load_2d(ldsA + nb * GEMM_TILE_B, Ag + (c + 1) * 64, 64, 128,
                  (unsigned long long)K, 64, 128, PAD_ROW128B);
      tdm_load_2d(ldsB + nb * GEMM_TILE_B, Bg + (c + 1) * 64, 64, 128,
                  (unsigned long long)K, 64, 128, PAD_ROW128B);
    }
    const __bf16* tA = lA + buf * (128 * STRIDE72);
    const __bf16* tB = lB + buf * (128 * STRIDE72);
#pragma unroll
    for (int ks = 0; ks < 64; ks += 32) {
      v16bf a = lds_a_frag(tA, STRIDE72, wave * 16, ks, lane);
      wmma_strip8(tB, STRIDE72, ks, lane, a, acc);
    }
  }
}

// ---------------------------------------------------------------------------
// QKV GEMM + fused QK RMSNorm epilogue (N-tile == one 128-wide head slice).
__global__ void __launch_bounds__(256, 2)
qkv_gemm_norm_kernel(const __bf16* __restrict__ x, const __bf16* __restrict__ w,
                     const float* __restrict__ q_scale,
                     const float* __restrict__ k_scale,
                     __bf16* __restrict__ qn, __bf16* __restrict__ kn,
                     __bf16* __restrict__ vT) {
  __shared__ __align__(32) char smem[4 * GEMM_TILE_B];
  const int lane = threadIdx.x & 31;
  const int wave = threadIdx.x >> 5;
  const int m0 = blockIdx.x * 128 + wave * 16;
  const int n0 = blockIdx.y * 128;

  v8f acc[8] = {};
  gemm_mainloop_tdm(x + (size_t)blockIdx.x * 128 * BDIM,
                    w + (size_t)n0 * BDIM, BDIM, smem, wave, lane, acc);

  const int n     = lane & 15;
  const int rbase = (lane >> 4) << 3;
  const int which = n0 / BDIM;             // 0=q, 1=k, 2=v
  const int h     = (n0 % BDIM) / HD;

  if (which == 2) {
#pragma unroll
    for (int r = 0; r < 8; ++r) {
      const int l  = m0 + rbase + r;       // 0..BL-1
      const int bh = (l >> 11) * NH + h;
      const int li = l & (SEQ - 1);
#pragma unroll
      for (int f = 0; f < 8; ++f) {
        const int hd = f * 16 + n;
        vT[((size_t)bh * HD + hd) * SEQ + li] = (__bf16)acc[f][r];
      }
    }
  } else {
    const float* scale = (which == 0) ? q_scale : k_scale;
    const float extra  = (which == 0) ? 0.08838834764831845f : 1.0f; // 1/sqrt(128)
    __bf16* dst        = (which == 0) ? qn : kn;
#pragma unroll
    for (int r = 0; r < 8; ++r) {
      float ss = 0.f;
#pragma unroll
      for (int f = 0; f < 8; ++f) ss += acc[f][r] * acc[f][r];
#pragma unroll
      for (int off = 1; off < 16; off <<= 1) ss += __shfl_xor(ss, off, 16);
      const float rrms = rsqrtf(ss * (1.0f / 128.0f) + 1e-6f) * extra;
      const int l  = m0 + rbase + r;
      const int bh = (l >> 11) * NH + h;
      const int li = l & (SEQ - 1);
      __bf16* row = dst + ((size_t)bh * SEQ + li) * HD;
#pragma unroll
      for (int f = 0; f < 8; ++f) {
        const int hd = f * 16 + n;
        row[hd] = (__bf16)(acc[f][r] * rrms * scale[hd]);
      }
    }
  }
}

// ---------------------------------------------------------------------------
// Flash attention. Block = 8 waves sharing one (b,h); wave w owns q-tile
// qgroup*8+w. K/V streamed in 64-key chunks via TDM into shared LDS tiles.
#define ATT_KT_B (64 * STRIDE136 * 2)   // 17408 bytes per K tile
#define ATT_VT_B (128 * STRIDE72 * 2)   // 18432 bytes per V tile

__global__ void __launch_bounds__(256, 2)
attn_kernel(const __bf16* __restrict__ q, const __bf16* __restrict__ k,
            const __bf16* __restrict__ vT, __bf16* __restrict__ attn_out) {
  __shared__ __align__(32) char smem[2 * ATT_KT_B + 2 * ATT_VT_B + 8 * 1024];
  __bf16* lK = (__bf16*)smem;                               // [2][64*136]
  __bf16* lV = (__bf16*)(smem + 2 * ATT_KT_B);              // [2][128*72]
  __bf16* lP = (__bf16*)(smem + 2 * ATT_KT_B + 2 * ATT_VT_B); // [8][16*32]
  const unsigned ldsK = (unsigned)(size_t)(void*)lK;
  const unsigned ldsV = (unsigned)(size_t)(void*)lV;

  const int lane = threadIdx.x & 31;
  const int wave = threadIdx.x >> 5;
  const int bh   = blockIdx.x >> 4;             // 0..47
  const int qt   = (blockIdx.x & 15) * 8 + wave; // q-tile in sequence (0..127)

  const __bf16* qp = q  + ((size_t)bh * SEQ + qt * 16) * HD;
  const __bf16* kp = k  + (size_t)bh * SEQ * HD;
  const __bf16* vp = vT + (size_t)bh * HD * SEQ;

  v16bf qf[4];
#pragma unroll
  for (int kk = 0; kk < 4; ++kk) qf[kk] = load_a16(qp, HD, 0, kk * 32, lane);

  v8f acc[8] = {};
  float mrow[8], lrow[8];
#pragma unroll
  for (int r = 0; r < 8; ++r) { mrow[r] = -1e30f; lrow[r] = 0.f; }

  __bf16* myp = lP + wave * (16 * 32);
  const int n     = lane & 15;
  const int rbase = (lane >> 4) << 3;
  const int NC = SEQ / 64;  // 32 chunks

  if (wave == 0) {
    tdm_load_2d(ldsK, kp, HD, 64, HD, HD, 64, PAD_ROW256B);        // 64 k-rows
    tdm_load_2d(ldsV, vp, 64, HD, SEQ, 64, HD, PAD_ROW128B);       // 128 hd-rows
  }
  for (int c = 0; c < NC; ++c) {
    const int buf = c & 1;
    if (wave == 0) __builtin_amdgcn_s_wait_tensorcnt((short)0);
    __syncthreads();
    if (wave == 0 && c + 1 < NC) {
      const int nb = buf ^ 1;
      tdm_load_2d(ldsK + nb * ATT_KT_B, kp + (size_t)(c + 1) * 64 * HD,
                  HD, 64, HD, HD, 64, PAD_ROW256B);
      tdm_load_2d(ldsV + nb * ATT_VT_B, vp + (c + 1) * 64,
                  64, HD, SEQ, 64, HD, PAD_ROW128B);
    }
    const __bf16* tK = lK + buf * (64 * STRIDE136);
    const __bf16* tV = lV + buf * (128 * STRIDE72);

#pragma unroll
    for (int sub = 0; sub < 2; ++sub) {          // two 32-key halves
      // ---- S = Q @ K^T (16x32), depth-2 pipelined over 8 fragments ----
      // fragment j: columns = keys sub*32 + (j&1)*16, K-dim = hd (j>>1)*32
      v8f s0 = {}, s1 = {};
      {
        v16bf b0 = lds_b_frag(tK, STRIDE136, sub * 32,      0, lane);
        v16bf b1 = lds_b_frag(tK, STRIDE136, sub * 32 + 16, 0, lane);
        v16bf b2;
#pragma unroll
        for (int j = 0; j < 8; ++j) {
          if (j + 2 < 8)
            b2 = lds_b_frag(tK, STRIDE136, sub * 32 + ((j + 2) & 1) * 16,
                            ((j + 2) >> 1) * 32, lane);
          if ((j & 1) == 0) s0 = wmma_bf16(qf[j >> 1], b0, s0);
          else              s1 = wmma_bf16(qf[j >> 1], b0, s1);
          b0 = b1;
          b1 = b2;
        }
      }
      // ---- online softmax ----
      float alpha[8];
#pragma unroll
      for (int r = 0; r < 8; ++r) {
        float t = fmaxf(s0[r], s1[r]);
#pragma unroll
        for (int off = 1; off < 16; off <<= 1) t = fmaxf(t, __shfl_xor(t, off, 16));
        const float mnew = fmaxf(mrow[r], t);
        alpha[r] = __expf(mrow[r] - mnew);
        const float p0 = __expf(s0[r] - mnew);
        const float p1 = __expf(s1[r] - mnew);
        myp[(rbase + r) * 32 + n]      = (__bf16)p0;  // D->A relayout via LDS
        myp[(rbase + r) * 32 + 16 + n] = (__bf16)p1;
        float ps = p0 + p1;
#pragma unroll
        for (int off = 1; off < 16; off <<= 1) ps += __shfl_xor(ps, off, 16);
        lrow[r] = lrow[r] * alpha[r] + ps;
        mrow[r] = mnew;
      }
#pragma unroll
      for (int f = 0; f < 8; ++f)
#pragma unroll
        for (int r = 0; r < 8; ++r) acc[f][r] *= alpha[r];

      // ---- O += P @ V (pf read after same-wave P stores; DS is in-order) ----
      v16bf pf = lds_a_frag(myp, 32, 0, 0, lane);
      wmma_strip8(tV, STRIDE72, sub * 32, lane, pf, acc);
    }
  }

  // ---- normalize and store to merged-head layout [B, L, D] (bf16) ----
  const int b = bh / NH, h = bh % NH;
#pragma unroll
  for (int r = 0; r < 8; ++r) {
    const float inv = 1.0f / lrow[r];
    const int row = qt * 16 + rbase + r;
    __bf16* orow = attn_out + ((size_t)b * SEQ + row) * BDIM + h * HD;
#pragma unroll
    for (int f = 0; f < 8; ++f) orow[f * 16 + n] = (__bf16)(acc[f][r] * inv);
  }
}

// ---------------------------------------------------------------------------
// Output projection GEMM: [BL,D] x [D,D]^T -> fp32 out.
__global__ void __launch_bounds__(256, 2)
proj_gemm_kernel(const __bf16* __restrict__ a, const __bf16* __restrict__ w,
                 float* __restrict__ out) {
  __shared__ __align__(32) char smem[4 * GEMM_TILE_B];
  const int lane = threadIdx.x & 31;
  const int wave = threadIdx.x >> 5;
  const int m0 = blockIdx.x * 128 + wave * 16;
  const int n0 = blockIdx.y * 128;

  v8f acc[8] = {};
  gemm_mainloop_tdm(a + (size_t)blockIdx.x * 128 * BDIM,
                    w + (size_t)n0 * BDIM, BDIM, smem, wave, lane, acc);

  const int n     = lane & 15;
  const int rbase = (lane >> 4) << 3;
#pragma unroll
  for (int r = 0; r < 8; ++r) {
    float* orow = out + (size_t)(m0 + rbase + r) * BDIM + n0;
#pragma unroll
    for (int f = 0; f < 8; ++f) orow[f * 16 + n] = acc[f][r];
  }
}

// ---------------------------------------------------------------------------
extern "C" void kernel_launch(void* const* d_in, const int* in_sizes, int n_in,
                              void* d_out, int out_size, void* d_ws, size_t ws_size,
                              hipStream_t stream) {
  (void)in_sizes; (void)n_in; (void)out_size; (void)ws_size;
  const float* x    = (const float*)d_in[0];  // [2,2048,3072]
  const float* qkvw = (const float*)d_in[1];  // [9216,3072]
  const float* qs   = (const float*)d_in[2];  // [128]
  const float* ks   = (const float*)d_in[3];  // [128]
  const float* pw   = (const float*)d_in[4];  // [3072,3072]
  float* out = (float*)d_out;

  char* ws = (char*)d_ws;
  size_t off = 0;
  auto take = [&](size_t bytes) {
    void* p = ws + off;
    off += (bytes + 255) & ~(size_t)255;
    return p;
  };
  __bf16* xb   = (__bf16*)take((size_t)BL * BDIM * 2);     // x bf16
  __bf16* wb   = (__bf16*)take((size_t)TD * BDIM * 2);     // qkv_w bf16
  __bf16* pb   = (__bf16*)take((size_t)BDIM * BDIM * 2);   // proj_w bf16
  __bf16* qn   = (__bf16*)take((size_t)BL * BDIM * 2);     // normalized q [bh][l][hd]
  __bf16* kn   = (__bf16*)take((size_t)BL * BDIM * 2);     // normalized k [bh][l][hd]
  __bf16* vTb  = (__bf16*)take((size_t)BL * BDIM * 2);     // v transposed [bh][hd][l]
  __bf16* attb = (__bf16*)take((size_t)BL * BDIM * 2);     // attention out [B,L,D]

  const int nx = BL * BDIM, nw = TD * BDIM, np = BDIM * BDIM;
  f32_to_bf16_kernel<<<(nx + 255) / 256, 256, 0, stream>>>(x, xb, nx);
  f32_to_bf16_kernel<<<(nw + 255) / 256, 256, 0, stream>>>(qkvw, wb, nw);
  f32_to_bf16_kernel<<<(np + 255) / 256, 256, 0, stream>>>(pw, pb, np);

  qkv_gemm_norm_kernel<<<dim3(BL / 128, TD / 128), 256, 0, stream>>>(
      xb, wb, qs, ks, qn, kn, vTb);

  attn_kernel<<<48 * 16, 256, 0, stream>>>(qn, kn, vTb, attb);

  proj_gemm_kernel<<<dim3(BL / 128, BDIM / 128), 256, 0, stream>>>(attb, pb, out);
}
